// MicroMoEFFN_21973052686456
// MI455X (gfx1250) — matrix operand order repaired and checked by reading, not compiled, not run on hardware
//
#include <hip/hip_runtime.h>
#include <hip/hip_bf16.h>

#define DK   1024   // model dim
#define FK   2048   // expert hidden dim
#define NTOK 8192   // B*T tokens
#define NEXP 2

typedef __attribute__((ext_vector_type(16))) __bf16 v16bf;
typedef __attribute__((ext_vector_type(8)))  __bf16 v8bf;
typedef __attribute__((ext_vector_type(8)))  float  v8f;
typedef int v4i_t __attribute__((vector_size(16)));   // matches builtin pointee

union FragU { v16bf v; v8bf h[2]; };

#ifndef __has_builtin
#define __has_builtin(x) 0
#endif

#if __has_builtin(__builtin_amdgcn_global_load_async_to_lds_b128)
#define USE_ASYNC 1
#else
#define USE_ASYNC 0
#endif

#if __has_builtin(__builtin_amdgcn_s_wait_asynccnt)
#define WAIT_ASYNC0() __builtin_amdgcn_s_wait_asynccnt(0)
#elif defined(__gfx1250__)
#define WAIT_ASYNC0() asm volatile("s_wait_asynccnt 0x0" ::: "memory")
#else
#define WAIT_ASYNC0()
#endif

#define GLB_AS __attribute__((address_space(1)))
#define LDS_AS __attribute__((address_space(3)))

#if USE_ASYNC
// Async DMA global->LDS, tracked by ASYNCcnt. Flat->local is a low-32-bit
// truncation (ISA aperture rules), flat global value == 64-bit VA, so
// integer-cast pointers carry the correct hardware addresses.
__device__ inline void async_b128(const __bf16* g, __bf16* l) {
  __builtin_amdgcn_global_load_async_to_lds_b128(
      (GLB_AS v4i_t*)(__UINTPTR_TYPE__)g,
      (LDS_AS v4i_t*)(unsigned int)(__UINTPTR_TYPE__)l, 0, 0);
}
#endif

// copy 16 bf16 (32B) global -> LDS
__device__ inline void copy16(const __bf16* s, __bf16* d) {
#if USE_ASYNC
  async_b128(s, d);
  async_b128(s + 8, d + 8);
#else
  const uint4 p0 = *(const uint4*)(s);
  const uint4 p1 = *(const uint4*)(s + 8);
  *(uint4*)(d) = p0;
  *(uint4*)(d + 8) = p1;
  __builtin_prefetch(s + 32, 0, 0);
#endif
}
// copy 8 bf16 (16B) global -> LDS
__device__ inline void copy8(const __bf16* s, __bf16* d) {
#if USE_ASYNC
  async_b128(s, d);
#else
  const uint4 p0 = *(const uint4*)(s);
  *(uint4*)(d) = p0;
  __builtin_prefetch(s + 32, 0, 0);
#endif
}

// ---------------------------------------------------------------------------
// LDS tile layout: row-major, 32 bf16 of K per row, padded to 40 bf16 (80 B)
// -> 16B-aligned fragment chunks, conflict-free bank rotation (80B = 20 dwords)
// ---------------------------------------------------------------------------

// A-matrix 16x32 bf16 fragment (ISA 7.12.2): lanes 0-15 rows, half=lane>>4
// selects K chunks {8h..8h+7} (v0-3) and {16+8h..16+8h+7} (v4-7).
__device__ inline v16bf load_frag_a(const __bf16* base, int rowTile, int lane) {
  const int r    = rowTile + (lane & 15);
  const int half = lane >> 4;
  const __bf16* p = base + r * 40 + half * 8;
  FragU f;
  f.h[0] = *(const v8bf*)(p);
  f.h[1] = *(const v8bf*)(p + 16);
  return f.v;
}

// B-matrix 32x16 bf16 fragment: lane holds column n=lane&15, K chunks
// {8kh..8kh+7} and {16+8kh..16+8kh+7} (kh = lane>>4); weights stored [outcol][K].
__device__ inline v16bf load_frag_b(const __bf16* base, int colTile, int lane) {
  const int c  = colTile + (lane & 15);
  const int kh = lane >> 4;
  const __bf16* p = base + c * 40 + kh * 8;
  FragU f;
  f.h[0] = *(const v8bf*)(p);
  f.h[1] = *(const v8bf*)(p + 16);
  return f.v;
}

// ---------------------------------------------------------------------------
// Prep kernels
// ---------------------------------------------------------------------------
__global__ void zero_cnt_kernel(int* cnt) {
  if (threadIdx.x < NEXP) cnt[threadIdx.x] = 0;
}

__global__ __launch_bounds__(256) void router_kernel(
    const float* __restrict__ x, const float* __restrict__ Wr,
    const float* __restrict__ bias, int* __restrict__ perm,
    int* __restrict__ cnt) {
  const int lane = threadIdx.x & 31;
  const int n = blockIdx.x * 8 + (threadIdx.x >> 5);
  const float* xr = x + (size_t)n * DK;
  float s0 = 0.f, s1 = 0.f;
  for (int d = lane; d < DK; d += 32) {
    const float xv = xr[d];
    s0 += xv * Wr[d * 2 + 0];
    s1 += xv * Wr[d * 2 + 1];
  }
  for (int off = 16; off; off >>= 1) {
    s0 += __shfl_down(s0, off, 32);
    s1 += __shfl_down(s1, off, 32);
  }
  if (lane == 0) {
    const int e = (s1 + bias[1] > s0 + bias[0]) ? 1 : 0;  // argmax, tie -> 0
    const int pos = atomicAdd(&cnt[e], 1);
    perm[e * NTOK + pos] = n;
  }
}

__global__ __launch_bounds__(256) void cast_x_kernel(
    const float* __restrict__ x, __bf16* __restrict__ xb) {
  const size_t i = ((size_t)blockIdx.x * 256 + threadIdx.x) * 4;
  const float4 v = *(const float4*)(x + i);
  struct bf4 { __bf16 a, b, c, d; } o;
  o.a = (__bf16)v.x; o.b = (__bf16)v.y; o.c = (__bf16)v.z; o.d = (__bf16)v.w;
  *(bf4*)(xb + i) = o;
}

// src[K][Fo] f32 (row-major) -> dst[Fo][K] bf16 ; K,Fo multiples of 32
__global__ __launch_bounds__(256) void cast_transpose_kernel(
    const float* __restrict__ src, __bf16* __restrict__ dst, int K, int Fo) {
  __shared__ float tile[32][33];
  const int tx = threadIdx.x, ty = threadIdx.y;
  const int f0 = blockIdx.x * 32, k0 = blockIdx.y * 32;
#pragma unroll
  for (int j = 0; j < 4; ++j)
    tile[ty + j * 8][tx] = src[(size_t)(k0 + ty + j * 8) * Fo + f0 + tx];
  __syncthreads();
#pragma unroll
  for (int j = 0; j < 4; ++j)
    dst[(size_t)(f0 + ty + j * 8) * K + k0 + tx] = (__bf16)tile[tx][ty + j * 8];
}

// ---------------------------------------------------------------------------
// Grouped rows: expert-1 region starts at align128(cnt0) so no tile mixes
// experts; rows in the alignment gap are invalid (computed, never stored).
// ---------------------------------------------------------------------------
template <int GATHERED>
__device__ inline int setup_rows(const int* perm, const int* cnt, int r0,
                                 int tid, int* rowTok) {
  int expert = 0;
  if (GATHERED) {
    const int c0 = cnt[0], c1 = cnt[1];
    const int a1 = (c0 + 127) & ~127;
    expert = (r0 >= a1) ? 1 : 0;
    if (tid < 128) {
      const int r = r0 + tid;
      int tok = -1;
      if (r < c0) tok = perm[r];
      else if (r >= a1 && r < a1 + c1) tok = perm[NTOK + (r - a1)];
      rowTok[tid] = tok;
    }
  } else {
    if (tid < 128) rowTok[tid] = r0 + tid;
  }
  return expert;
}

// ---------------------------------------------------------------------------
// Up-projection: h = silu(x@Wg) * (x@Wu). Fused dual GEMM, bf16 WMMA.
// Block tile 128(M)x64(F), 8 waves (4x2), wave 32x32 per output.
// 2-deep LDS pipeline, async DMA staging, ONE barrier per K-step:
//   wait_async ; barrier ; stage(next) ; compute(cur)
// stage(next) at iter i overwrites the compute buffer of iter i-1, which every
// wave has finished (all waves passed barrier(i) => all finished compute(i-1)).
// ---------------------------------------------------------------------------
template <int GATHERED>
__global__ __launch_bounds__(256) void moe_up_kernel(
    const __bf16* __restrict__ xb,                   // [NTOK][DK]
    const __bf16* __restrict__ WgT,                  // [(E)*FK][DK]
    const __bf16* __restrict__ WuT,
    const int* __restrict__ perm, const int* __restrict__ cnt,
    __bf16* __restrict__ hbuf) {                     // [(NTOK+128)][FK]
  __shared__ alignas(16) __bf16 lA[2][128 * 40];
  __shared__ alignas(16) __bf16 lBg[2][64 * 40];
  __shared__ alignas(16) __bf16 lBu[2][64 * 40];
  __shared__ int rowTok[128];

  const int tid   = threadIdx.x;
  const int fbase = blockIdx.x * 64;
  const int r0    = blockIdx.y * 128;

  const int expert = setup_rows<GATHERED>(perm, cnt, r0, tid, rowTok);
  __syncthreads();

  const __bf16* Bg = WgT + ((size_t)expert * FK + fbase) * DK;
  const __bf16* Bu = WuT + ((size_t)expert * FK + fbase) * DK;

  const int lane = tid & 31;
  const int wave = tid >> 5;
  const int wm = wave >> 1;  // 0..3 (M, 32 rows)
  const int wn = wave & 1;   // 0..1 (F, 32 cols)

  // staging slots: A 128 rows x 2 halves, B 64 rows x 2 halves (Wg | Wu)
  const int ar = tid >> 1, ah = tid & 1;
  const int bs = tid & 127, br = bs >> 1, bh = bs & 1;
  const __bf16* bsrc = (tid < 128) ? Bg : Bu;
  const int myTok = rowTok[ar];
  const __bf16* asrc = xb + (size_t)(myTok < 0 ? 0 : myTok) * DK;

  v8f accG[2][2] = {};
  v8f accU[2][2] = {};

  auto stage = [&](int buf, int kk) {
    copy16(asrc + kk + ah * 16, &lA[buf][ar * 40 + ah * 16]);
    __bf16* bd = ((tid < 128) ? lBg[buf] : lBu[buf]) + br * 40 + bh * 16;
    copy16(bsrc + (size_t)br * DK + kk + bh * 16, bd);
  };

  stage(0, 0);  // prologue

  for (int kk = 0; kk < DK; kk += 32) {
    const int cur = (kk >> 5) & 1;
#if USE_ASYNC
    WAIT_ASYNC0();            // this wave's DMA into buf[cur] has landed
#endif
    __syncthreads();          // everyone's buf[cur] visible
    if (kk + 32 < DK) stage(cur ^ 1, kk + 32);

    const __bf16* bA = lA[cur];
    const v16bf fa0 = load_frag_a(bA, wm * 32 + 0, lane);
    const v16bf fa1 = load_frag_a(bA, wm * 32 + 16, lane);
    const v16bf fg0 = load_frag_b(lBg[cur], wn * 32 + 0, lane);
    const v16bf fg1 = load_frag_b(lBg[cur], wn * 32 + 16, lane);
    const v16bf fu0 = load_frag_b(lBu[cur], wn * 32 + 0, lane);
    const v16bf fu1 = load_frag_b(lBu[cur], wn * 32 + 16, lane);

    accG[0][0] = __builtin_amdgcn_wmma_f32_16x16x32_bf16(false, fa0, false, fg0, (short)0, accG[0][0], false, false);
    accG[0][1] = __builtin_amdgcn_wmma_f32_16x16x32_bf16(false, fa0, false, fg1, (short)0, accG[0][1], false, false);
    accG[1][0] = __builtin_amdgcn_wmma_f32_16x16x32_bf16(false, fa1, false, fg0, (short)0, accG[1][0], false, false);
    accG[1][1] = __builtin_amdgcn_wmma_f32_16x16x32_bf16(false, fa1, false, fg1, (short)0, accG[1][1], false, false);
    accU[0][0] = __builtin_amdgcn_wmma_f32_16x16x32_bf16(false, fa0, false, fu0, (short)0, accU[0][0], false, false);
    accU[0][1] = __builtin_amdgcn_wmma_f32_16x16x32_bf16(false, fa0, false, fu1, (short)0, accU[0][1], false, false);
    accU[1][0] = __builtin_amdgcn_wmma_f32_16x16x32_bf16(false, fa1, false, fu0, (short)0, accU[1][0], false, false);
    accU[1][1] = __builtin_amdgcn_wmma_f32_16x16x32_bf16(false, fa1, false, fu1, (short)0, accU[1][1], false, false);
  }

  // epilogue: C/D layout elem(vgpr j, lane) = (M = j + 8*(l>>4), N = l&15)
#pragma unroll
  for (int mi = 0; mi < 2; ++mi)
#pragma unroll
    for (int ni = 0; ni < 2; ++ni)
#pragma unroll
      for (int j = 0; j < 8; ++j) {
        const int lr = wm * 32 + mi * 16 + j + 8 * (lane >> 4);
        const int fc = fbase + wn * 32 + ni * 16 + (lane & 15);
        const float g = accG[mi][ni][j];
        const float u = accU[mi][ni][j];
        const float hv = (g / (1.0f + __expf(-g))) * u;
        if (rowTok[lr] >= 0)
          hbuf[(size_t)(r0 + lr) * FK + fc] = (__bf16)hv;
      }
}

// ---------------------------------------------------------------------------
// Down-projection: out[token][d] (=|+=) h @ WdT ; K = FK = 2048.
// Same pipeline; block tile 128x64, 8 waves 4x2, wave 32x32.
// ---------------------------------------------------------------------------
template <int GATHERED>
__global__ __launch_bounds__(256) void moe_down_kernel(
    const __bf16* __restrict__ hbuf,                 // [(NTOK+128)][FK]
    const __bf16* __restrict__ WdT,                  // [(E)*DK][FK]
    const int* __restrict__ perm, const int* __restrict__ cnt,
    float* __restrict__ out) {                       // [NTOK][DK]
  __shared__ alignas(16) __bf16 lA[2][128 * 40];
  __shared__ alignas(16) __bf16 lB[2][64 * 40];
  __shared__ int rowTok[128];

  const int tid   = threadIdx.x;
  const int dbase = blockIdx.x * 64;
  const int r0    = blockIdx.y * 128;

  const int expert = setup_rows<GATHERED>(perm, cnt, r0, tid, rowTok);
  __syncthreads();

  const __bf16* Bw = WdT + ((size_t)expert * DK + dbase) * FK;

  const int lane = tid & 31;
  const int wave = tid >> 5;
  const int wm = wave >> 1;
  const int wn = wave & 1;

  const int ar = tid >> 1, ah = tid & 1;   // A: 128 rows x 2 halves (16 elems)
  const int br = tid >> 2, bq = tid & 3;   // B: 64 rows x 4 quarters (8 elems)
  const __bf16* asrc = hbuf + (size_t)(r0 + ar) * FK;

  v8f acc[2][2] = {};

  auto stage = [&](int buf, int kk) {
    copy16(asrc + kk + ah * 16, &lA[buf][ar * 40 + ah * 16]);
    copy8(Bw + (size_t)br * FK + kk + bq * 8, &lB[buf][br * 40 + bq * 8]);
  };

  stage(0, 0);

  for (int kk = 0; kk < FK; kk += 32) {
    const int cur = (kk >> 5) & 1;
#if USE_ASYNC
    WAIT_ASYNC0();
#endif
    __syncthreads();
    if (kk + 32 < FK) stage(cur ^ 1, kk + 32);

    const v16bf fa0 = load_frag_a(lA[cur], wm * 32 + 0, lane);
    const v16bf fa1 = load_frag_a(lA[cur], wm * 32 + 16, lane);
    const v16bf fb0 = load_frag_b(lB[cur], wn * 32 + 0, lane);
    const v16bf fb1 = load_frag_b(lB[cur], wn * 32 + 16, lane);

    acc[0][0] = __builtin_amdgcn_wmma_f32_16x16x32_bf16(false, fa0, false, fb0, (short)0, acc[0][0], false, false);
    acc[0][1] = __builtin_amdgcn_wmma_f32_16x16x32_bf16(false, fa0, false, fb1, (short)0, acc[0][1], false, false);
    acc[1][0] = __builtin_amdgcn_wmma_f32_16x16x32_bf16(false, fa1, false, fb0, (short)0, acc[1][0], false, false);
    acc[1][1] = __builtin_amdgcn_wmma_f32_16x16x32_bf16(false, fa1, false, fb1, (short)0, acc[1][1], false, false);
  }

#pragma unroll
  for (int mi = 0; mi < 2; ++mi)
#pragma unroll
    for (int ni = 0; ni < 2; ++ni)
#pragma unroll
      for (int j = 0; j < 8; ++j) {
        const int lr = wm * 32 + mi * 16 + j + 8 * (lane >> 4);
        const int dc = dbase + wn * 32 + ni * 16 + (lane & 15);
        const int tok = rowTok[lr];
        if (tok >= 0) {
          const size_t o = (size_t)tok * DK + dc;
          const float v = acc[mi][ni][j];
          if (GATHERED) out[o] += v;   // one writer per token
          else          out[o] = v;    // shared pass initializes out
        }
      }
}

// ---------------------------------------------------------------------------
extern "C" void kernel_launch(void* const* d_in, const int* in_sizes, int n_in,
                              void* d_out, int out_size, void* d_ws,
                              size_t ws_size, hipStream_t stream) {
  (void)in_sizes; (void)n_in; (void)out_size; (void)ws_size;
  const float* x  = (const float*)d_in[0];
  const float* Wr = (const float*)d_in[1];
  const float* rb = (const float*)d_in[2];
  const float* Wg = (const float*)d_in[3];
  const float* Wu = (const float*)d_in[4];
  const float* Wd = (const float*)d_in[5];
  const float* Sg = (const float*)d_in[6];
  const float* Su = (const float*)d_in[7];
  const float* Sd = (const float*)d_in[8];
  float* out = (float*)d_out;

  char* p = (char*)d_ws;
  auto take = [&](size_t bytes) -> char* {
    char* r = p;
    p += (bytes + 255) & ~(size_t)255;
    return r;
  };
  __bf16* xb  = (__bf16*)take((size_t)NTOK * DK * 2);          // 16 MB
  __bf16* WgT = (__bf16*)take((size_t)NEXP * FK * DK * 2);     //  8 MB
  __bf16* WuT = (__bf16*)take((size_t)NEXP * FK * DK * 2);     //  8 MB
  __bf16* WdT = (__bf16*)take((size_t)NEXP * DK * FK * 2);     //  8 MB
  __bf16* SgT = (__bf16*)take((size_t)FK * DK * 2);            //  4 MB
  __bf16* SuT = (__bf16*)take((size_t)FK * DK * 2);            //  4 MB
  __bf16* SdT = (__bf16*)take((size_t)DK * FK * 2);            //  4 MB
  __bf16* hbuf = (__bf16*)take((size_t)(NTOK + 128) * FK * 2); // ~33 MB
  int* perm = (int*)take((size_t)2 * NTOK * 4);
  int* cnt  = (int*)take(256);

  // 1) routing
  zero_cnt_kernel<<<1, 32, 0, stream>>>(cnt);
  router_kernel<<<NTOK / 8, 256, 0, stream>>>(x, Wr, rb, perm, cnt);

  // 2) precision / layout conversion (weights -> transposed bf16 [out][K])
  cast_x_kernel<<<(NTOK * DK) / (256 * 4), 256, 0, stream>>>(x, xb);
  dim3 tb(32, 8);
  for (int e = 0; e < NEXP; ++e) {
    cast_transpose_kernel<<<dim3(FK / 32, DK / 32), tb, 0, stream>>>(
        Wg + (size_t)e * DK * FK, WgT + (size_t)e * FK * DK, DK, FK);
    cast_transpose_kernel<<<dim3(FK / 32, DK / 32), tb, 0, stream>>>(
        Wu + (size_t)e * DK * FK, WuT + (size_t)e * FK * DK, DK, FK);
    cast_transpose_kernel<<<dim3(DK / 32, FK / 32), tb, 0, stream>>>(
        Wd + (size_t)e * FK * DK, WdT + (size_t)e * DK * FK, FK, DK);
  }
  cast_transpose_kernel<<<dim3(FK / 32, DK / 32), tb, 0, stream>>>(Sg, SgT, DK, FK);
  cast_transpose_kernel<<<dim3(FK / 32, DK / 32), tb, 0, stream>>>(Su, SuT, DK, FK);
  cast_transpose_kernel<<<dim3(DK / 32, FK / 32), tb, 0, stream>>>(Sd, SdT, FK, DK);

  // 3) shared expert: out = (silu(x@Sg)*(x@Su)) @ Sd
  moe_up_kernel<0><<<dim3(FK / 64, NTOK / 128), 256, 0, stream>>>(
      xb, SgT, SuT, nullptr, nullptr, hbuf);
  moe_down_kernel<0><<<dim3(DK / 64, NTOK / 128), 256, 0, stream>>>(
      hbuf, SdT, nullptr, nullptr, out);

  // 4) routed expert (grouped rows): out += y[sel]
  const int mblocks = NTOK / 128 + 1;  // padded grouped space <= NTOK+128 rows
  moe_up_kernel<1><<<dim3(FK / 64, mblocks), 256, 0, stream>>>(
      xb, WgT, WuT, perm, cnt, hbuf);
  moe_down_kernel<1><<<dim3(DK / 64, mblocks), 256, 0, stream>>>(
      hbuf, WdT, perm, cnt, out);
}
